// ResidualBlock_64450279244495
// MI455X (gfx1250) — matrix operand order repaired and checked by reading, not compiled, not run on hardware
//
#include <hip/hip_runtime.h>
#include <hip/hip_bf16.h>

typedef __attribute__((ext_vector_type(16))) _Float16 v16h;
typedef __attribute__((ext_vector_type(8)))  float    v8f;

#define CCH   128            // channels
#define KOFF  27             // 3x3x3 offsets
#define WCHUNK 16384         // halves per k-offset of packed W (128x128)

union Frag16 { v16h v; float4 f[2]; };

// ---------------------------------------------------------------------------
// small utility kernels
// ---------------------------------------------------------------------------
__global__ void zero_stats_kernel(float* stats) {
    int i = blockIdx.x * blockDim.x + threadIdx.x;
    if (i < 1024) stats[i] = 0.0f;
}

__global__ void cast_x_kernel(const float* __restrict__ x, _Float16* __restrict__ xh, long total) {
    long i = (long)blockIdx.x * blockDim.x + threadIdx.x;
    if (i < total) xh[i] = (_Float16)x[i];
}

// Pack W[k][c_in][c_out] (f32) into WMMA B-fragment order (f16):
// p = (((k*4 + kc)*8 + ct)*32 + lane)*16 + h
// lane = output column (mod 16); halves h -> K rows: lanes<16 K=(h&7)+(h>=8?16:0), lanes>=16 K=8+(h&7)+(h>=8?16:0)
__global__ void pack_w_kernel(const float* __restrict__ W, _Float16* __restrict__ wp, int total) {
    int p = blockIdx.x * blockDim.x + threadIdx.x;
    if (p >= total) return;
    int h    = p & 15;
    int lane = (p >> 4) & 31;
    int ct   = (p >> 9) & 7;
    int kc   = (p >> 12) & 3;
    int k    = p >> 14;
    int kl   = ((lane & 16) ? 8 : 0) + ((h & 8) ? 16 : 0) + (h & 7);
    int cin  = kc * 32 + kl;
    int cout = ct * 16 + (lane & 15);
    wp[p] = (_Float16)W[((size_t)k * CCH + cin) * CCH + cout];
}

// ---------------------------------------------------------------------------
// core gather-GEMM conv:  y[i,:] = sum_k xh[nbr[i,k],:] @ Wk   (+ BN stats)
// block = 256 threads = 8 waves; wave handles 32 rows x 128 cols
// (two A fragments share every B fragment -> halves LDS B traffic per WMMA)
// ---------------------------------------------------------------------------
__global__ void __launch_bounds__(256)
conv_wmma_kernel(const _Float16* __restrict__ xh,
                 const int*      __restrict__ nbr,
                 const _Float16* __restrict__ wp,
                 float*          __restrict__ y,
                 float*          __restrict__ gsum,
                 float*          __restrict__ gsq,
                 int Nv)
{
    __shared__ _Float16 sB[WCHUNK];     // 32 KB: W[k] staged per iteration
    __shared__ float    sStat[256];     // [0..127]=sum, [128..255]=sumsq

    const int tid  = threadIdx.x;
    const int lane = tid & 31;
    const int wave = tid >> 5;
    const int rowBase = blockIdx.x * 256 + wave * 32;

    sStat[tid] = 0.0f;                  // visible after first loop barrier

    v8f zero = {};
    v8f acc[2][8];
#pragma unroll
    for (int m = 0; m < 2; ++m)
#pragma unroll
        for (int t = 0; t < 8; ++t) acc[m][t] = zero;

    const int row0 = rowBase + (lane & 15);
    const int row1 = row0 + 16;
    const int rc0  = row0 < Nv ? row0 : (Nv - 1);
    const int rc1  = row1 < Nv ? row1 : (Nv - 1);
    const long nb0 = (long)rc0 * KOFF;
    const long nb1 = (long)rc1 * KOFF;
    const int aoff = (lane & 16) ? 8 : 0;

    for (int k = 0; k < KOFF; ++k) {
        __syncthreads();   // protect previous-iteration LDS reads (also fences sStat init)
        {   // cooperative stage of W[k] fragment block: 2048 float4
            const float4* src = reinterpret_cast<const float4*>(wp + (size_t)k * WCHUNK);
            float4* dst = reinterpret_cast<float4*>(sB);
#pragma unroll
            for (int i = 0; i < 8; ++i) dst[i * 256 + tid] = src[i * 256 + tid];
            if (k + 1 < KOFF)  // prefetch next chunk (global_prefetch_b8), 256 x 128B = 32KB
                __builtin_prefetch(reinterpret_cast<const char*>(wp + (size_t)(k + 1) * WCHUNK)
                                   + tid * 128, 0, 1);
        }
        __syncthreads();

        const int g0 = nbr[nb0 + k];
        const int g1 = nbr[nb1 + k];
        const _Float16* xr0 = xh + (size_t)g0 * CCH;
        const _Float16* xr1 = xh + (size_t)g1 * CCH;

#pragma unroll
        for (int kc = 0; kc < 4; ++kc) {
            Frag16 a0, a1;   // A frags: ISA 16-bit A layout via two 16B loads per lane
            a0.f[0] = *reinterpret_cast<const float4*>(xr0 + kc * 32 + aoff);
            a0.f[1] = *reinterpret_cast<const float4*>(xr0 + kc * 32 + aoff + 16);
            a1.f[0] = *reinterpret_cast<const float4*>(xr1 + kc * 32 + aoff);
            a1.f[1] = *reinterpret_cast<const float4*>(xr1 + kc * 32 + aoff + 16);
#pragma unroll
            for (int ct = 0; ct < 8; ++ct) {
                Frag16 b;
                const float4* bp = reinterpret_cast<const float4*>(
                    sB + ((kc * 8 + ct) * 512 + lane * 16));
                b.f[0] = bp[0];
                b.f[1] = bp[1];
                acc[0][ct] = __builtin_amdgcn_wmma_f32_16x16x32_f16(
                    false, a0.v, false, b.v, (short)0, acc[0][ct], false, false);
                acc[1][ct] = __builtin_amdgcn_wmma_f32_16x16x32_f16(
                    false, a1.v, false, b.v, (short)0, acc[1][ct], false, false);
            }
        }
    }

    // epilogue: store y + per-channel sum/sumsq (register pre-reduce -> LDS -> global)
#pragma unroll
    for (int ct = 0; ct < 8; ++ct) {
        float s = 0.0f, sq = 0.0f;
        const int col = ct * 16 + (lane & 15);
#pragma unroll
        for (int m = 0; m < 2; ++m) {
#pragma unroll
            for (int v = 0; v < 8; ++v) {
                const int row = rowBase + m * 16 + v + ((lane & 16) ? 8 : 0);
                float val = acc[m][ct][v];
                if (row < Nv) {
                    y[(size_t)row * CCH + col] = val;
                    s  += val;
                    sq += val * val;
                }
            }
        }
        atomicAdd(&sStat[col], s);
        atomicAdd(&sStat[128 + col], sq);
    }
    __syncthreads();
    if (tid < 128)      atomicAdd(&gsum[tid], sStat[tid]);
    else                atomicAdd(&gsq[tid - 128], sStat[tid]);
}

// ---------------------------------------------------------------------------
// BN finalize + elementwise kernels
// ---------------------------------------------------------------------------
__global__ void bn_finalize_kernel(const float* __restrict__ gsum, const float* __restrict__ gsq,
                                   const float* __restrict__ g, const float* __restrict__ b,
                                   float* __restrict__ scale, float* __restrict__ shift, int Nv)
{
    int c = threadIdx.x;
    if (c >= CCH) return;
    float inv = 1.0f / (float)Nv;
    float m   = gsum[c] * inv;
    float var = gsq[c] * inv - m * m;
    float sc  = g[c] * rsqrtf(var + 1e-5f);
    scale[c] = sc;
    shift[c] = b[c] - m * sc;
}

__global__ void bn_relu_cast_kernel(const float* __restrict__ y,
                                    const float* __restrict__ scale,
                                    const float* __restrict__ shift,
                                    _Float16* __restrict__ h, long total)
{
    long i = (long)blockIdx.x * blockDim.x + threadIdx.x;
    if (i >= total) return;
    int c = (int)(i & (CCH - 1));
    float v = y[i] * scale[c] + shift[c];
    h[i] = (_Float16)(v > 0.0f ? v : 0.0f);
}

__global__ void final_kernel(const float* __restrict__ y,
                             const float* __restrict__ scale,
                             const float* __restrict__ shift,
                             const float* __restrict__ x,
                             float* __restrict__ out, long total)
{
    long i = (long)blockIdx.x * blockDim.x + threadIdx.x;
    if (i >= total) return;
    int c = (int)(i & (CCH - 1));
    float v = y[i] * scale[c] + shift[c] + x[i];
    out[i] = v > 0.0f ? v : 0.0f;
}

// ---------------------------------------------------------------------------
extern "C" void kernel_launch(void* const* d_in, const int* in_sizes, int n_in,
                              void* d_out, int out_size, void* d_ws, size_t ws_size,
                              hipStream_t stream)
{
    const float* x   = (const float*)d_in[0];
    const int*   nbr = (const int*)  d_in[1];
    const float* W1  = (const float*)d_in[2];
    const float* g1  = (const float*)d_in[3];
    const float* b1  = (const float*)d_in[4];
    const float* W2  = (const float*)d_in[5];
    const float* g2  = (const float*)d_in[6];
    const float* b2  = (const float*)d_in[7];
    float* out = (float*)d_out;

    const int  Nv = in_sizes[0] / CCH;
    const long nc = (long)Nv * CCH;

    // workspace carve-up (reused buffers: xh holds x_f16 then h1_f16; y reused by both convs)
    char* ws = (char*)d_ws;
    _Float16* xh  = (_Float16*)ws;                              // nc halves
    float*    y   = (float*)(ws + nc * 2);                      // nc floats
    _Float16* wp1 = (_Float16*)(ws + nc * 2 + nc * 4);          // KOFF*WCHUNK halves
    _Float16* wp2 = wp1 + (size_t)KOFF * WCHUNK;
    float* stats  = (float*)(wp2 + (size_t)KOFF * WCHUNK);      // 1024 floats
    float *sum1 = stats,       *sq1 = stats + 128, *sc1 = stats + 256, *sh1 = stats + 384;
    float *sum2 = stats + 512, *sq2 = stats + 640, *sc2 = stats + 768, *sh2 = stats + 896;

    const int elemBlocks = (int)((nc + 255) / 256);
    const int convBlocks = (Nv + 255) / 256;
    const int packTotal  = KOFF * WCHUNK;
    const int packBlocks = (packTotal + 255) / 256;

    zero_stats_kernel<<<4, 256, 0, stream>>>(stats);
    cast_x_kernel<<<elemBlocks, 256, 0, stream>>>(x, xh, nc);
    pack_w_kernel<<<packBlocks, 256, 0, stream>>>(W1, wp1, packTotal);
    pack_w_kernel<<<packBlocks, 256, 0, stream>>>(W2, wp2, packTotal);

    // conv1 + BN1 stats
    conv_wmma_kernel<<<convBlocks, 256, 0, stream>>>(xh, nbr, wp1, y, sum1, sq1, Nv);
    bn_finalize_kernel<<<1, 128, 0, stream>>>(sum1, sq1, g1, b1, sc1, sh1, Nv);
    bn_relu_cast_kernel<<<elemBlocks, 256, 0, stream>>>(y, sc1, sh1, xh, nc); // xh := h1 (f16)

    // conv2 + BN2 stats (y overwritten)
    conv_wmma_kernel<<<convBlocks, 256, 0, stream>>>(xh, nbr, wp2, y, sum2, sq2, Nv);
    bn_finalize_kernel<<<1, 128, 0, stream>>>(sum2, sq2, g2, b2, sc2, sh2, Nv);

    // out = relu(bn2(y) + x)
    final_kernel<<<elemBlocks, 256, 0, stream>>>(y, sc2, sh2, x, out, nc);
}
